// A3TGCNWithMap_23776938951052
// MI455X (gfx1250) — matrix operand order repaired and checked by reading, not compiled, not run on hardware
//
#include <hip/hip_runtime.h>
#include <hip/hip_bf16.h>

typedef __bf16 bf16;
typedef __attribute__((ext_vector_type(16))) __bf16 v16bf;
typedef __attribute__((ext_vector_type(8)))  __bf16 v8bf;
typedef __attribute__((ext_vector_type(8)))  float  v8f;

#ifndef __HIP_DEVICE_COMPILE__
#endif

// ---- fragment pool layout: each frag = 32 lanes * 16 bf16 = 512 elems ----
#define F_WZC 0    // Wz_c  32x32 -> 2 frags (nb=0,1)
#define F_WHC 2    // Wh_c  32x32 -> 2 frags
#define F_WZL 4    // Wz_l[0:32,:] 32x32 -> 2 frags
#define F_WHL 6    // Wh_l[0:32,:] 32x32 -> 2 frags
#define F_W1  8    // W1 32x64 -> 4 frags
#define F_W2  12   // W2 64x112(pad from 100) -> 2 ktiles x 7 ntiles = 14 frags
#define NFRAG 26

// ---------------- WMMA helpers (CDNA5 16x16x32 bf16, f32 accum) ----------------
static __device__ __forceinline__ v16bf load_A16x32(const bf16* X, int row0, int ld,
                                                    int kbase, int lane) {
  // A k order used here (contiguous, matches B frags): lane<16 -> M=lane, K = kbase + {0..15}
  //                  lane>=16 -> M=lane-16, K = kbase + {16..31}
  const int half = lane >> 4;
  const bf16* r = X + (size_t)(row0 + (lane & 15)) * ld + (kbase + half * 16);
  v8bf lo = *(const v8bf*)(r);        // 16B load
  v8bf hi = *(const v8bf*)(r + 8);    // 16B load
  v16bf a;
#pragma unroll
  for (int i = 0; i < 8; ++i) { a[i] = lo[i]; a[i + 8] = hi[i]; }
  return a;
}

static __device__ __forceinline__ v16bf load_B(const bf16* pool, int frag, int lane) {
  return *(const v16bf*)(pool + ((size_t)frag * 32 + lane) * 16);  // 32B contiguous
}

static __device__ __forceinline__ v8f wmma_bf16(v16bf a, v16bf b, v8f c) {
  return __builtin_amdgcn_wmma_f32_16x16x32_bf16(false, a, false, b, (short)0, c,
                                                 false, false);
}

// ---------------- tiny scalar: s = sum(softmax(attn)) ----------------
__global__ void k_attn(const float* attn, int P, float* s_out) {
  if (threadIdx.x == 0 && blockIdx.x == 0) {
    float mx = attn[0];
    for (int i = 1; i < P; ++i) mx = fmaxf(mx, attn[i]);
    float den = 0.f;
    for (int i = 0; i < P; ++i) den += __expf(attn[i] - mx);
    float tot = 0.f;
    for (int i = 0; i < P; ++i) tot += __expf(attn[i] - mx) / den;
    *s_out = tot;
  }
}

// ---------------- weight pre-swizzle into WMMA B layout (bf16) ----------------
// B frag element: lane l, elem e  ->  K = kb*32 + 16*(l>>4) + e,  N = nb*16 + (l&15)
__global__ void k_prep(const float* Wzc, const float* Whc, const float* Wzl,
                       const float* Whl, const float* W1, const float* W2,
                       bf16* pool) {
  int t = blockIdx.x * blockDim.x + threadIdx.x;
  if (t >= NFRAG * 512) return;
  int f = t >> 9, r = t & 511, l = r >> 4, e = r & 15;
  const float* W; int ldn, kb, nb, nlim;
  if (f < 2)       { W = Wzc; ldn = 32;  kb = 0;          nb = f;          nlim = 32; }
  else if (f < 4)  { W = Whc; ldn = 32;  kb = 0;          nb = f - 2;      nlim = 32; }
  else if (f < 6)  { W = Wzl; ldn = 32;  kb = 0;          nb = f - 4;      nlim = 32; }
  else if (f < 8)  { W = Whl; ldn = 32;  kb = 0;          nb = f - 6;      nlim = 32; }
  else if (f < 12) { W = W1;  ldn = 64;  kb = 0;          nb = f - 8;      nlim = 64; }
  else             { W = W2;  ldn = 100; kb = (f - 12) / 7; nb = (f - 12) % 7; nlim = 100; }
  int k = kb * 32 + 16 * (l >> 4) + e;
  int n = nb * 16 + (l & 15);
  float v = (n < nlim) ? W[(size_t)k * ldn + n] : 0.0f;
  pool[t] = (bf16)v;
}

// ---------------- encoders -> x_bf [N,32] bf16 row-major ----------------
__global__ void k_encode(const float* ax, const float* mx, const float* Wa,
                         const float* ba, const float* Wm, const float* bm,
                         bf16* x_bf, int NA, int NM) {
  int t = blockIdx.x * blockDim.x + threadIdx.x;
  int Ntot = NA + NM;
  if (t >= Ntot * 32) return;
  int i = t >> 5, c = t & 31;
  float acc;
  if (i < NA) {
    acc = ba[c];
#pragma unroll
    for (int j = 0; j < 9; ++j) acc += ax[(size_t)i * 9 + j] * Wa[j * 32 + c];
  } else {
    int im = i - NA;
    acc = bm[c];
#pragma unroll
    for (int j = 0; j < 6; ++j) acc += mx[(size_t)im * 6 + j] * Wm[j * 32 + c];
  }
  x_bf[t] = (bf16)acc;
}

// ---------------- degree / norm ----------------
__global__ void k_initdeg(float* deg, int N) {
  int i = blockIdx.x * blockDim.x + threadIdx.x;
  if (i < N) deg[i] = 1.0f;  // self loop
}
__global__ void k_deg(const int* ei, int E, float* deg) {
  int e = blockIdx.x * blockDim.x + threadIdx.x;
  if (e < E) unsafeAtomicAdd(&deg[ei[(size_t)E + e]], 1.0f);
}
__global__ void k_dinv(float* dinv, int N) {
  int i = blockIdx.x * blockDim.x + threadIdx.x;
  if (i < N) { float d = dinv[i]; dinv[i] = d > 0.f ? rsqrtf(d) : 0.f; }
}

__global__ void k_zero2(float* a, float* b, int n) {
  int i = blockIdx.x * blockDim.x + threadIdx.x;
  if (i < n) { a[i] = 0.f; b[i] = 0.f; }
}

// ---------------- xz = x@Wz_c, xh = x@Wh_c via WMMA ----------------
__global__ void k_xw(const bf16* x_bf, const bf16* pool, float* xz, float* xh,
                     int ntiles) {
  int wave = blockIdx.x * (blockDim.x >> 5) + (threadIdx.x >> 5);
  if (wave >= ntiles) return;
  int lane = threadIdx.x & 31;
  v16bf a = load_A16x32(x_bf, wave * 16, 32, 0, lane);
  int nn = lane & 15, mb = 8 * (lane >> 4);
#pragma unroll
  for (int nb = 0; nb < 2; ++nb) {
    v8f cz = {}; v8f ch = {};
    cz = wmma_bf16(a, load_B(pool, F_WZC + nb, lane), cz);
    ch = wmma_bf16(a, load_B(pool, F_WHC + nb, lane), ch);
#pragma unroll
    for (int v = 0; v < 8; ++v) {
      size_t o = (size_t)(wave * 16 + mb + v) * 32 + nb * 16 + nn;
      xz[o] = cz[v];
      xh[o] = ch[v];
    }
  }
}

// ---------------- edge scatter: az/ah += norm * xz/xh[src] (L2-resident) ----------------
__global__ void k_edge(const int* ei, int E, const float* dinv, const float* xz,
                       const float* xh, float* az, float* ah) {
  int t = blockIdx.x * blockDim.x + threadIdx.x;
  if (t >= E * 8) return;
  int e = t >> 3, q = t & 7;             // 8 lanes share one edge -> coalesced rows
  int s = ei[e], d = ei[(size_t)E + e];
  float nrm = dinv[s] * dinv[d];
  size_t so = (size_t)s * 32 + q * 4;
  size_t doff = (size_t)d * 32 + q * 4;
  const float4 vz = *(const float4*)(xz + so);
  const float4 vh = *(const float4*)(xh + so);
  unsafeAtomicAdd(&az[doff + 0], vz.x * nrm);
  unsafeAtomicAdd(&az[doff + 1], vz.y * nrm);
  unsafeAtomicAdd(&az[doff + 2], vz.z * nrm);
  unsafeAtomicAdd(&az[doff + 3], vz.w * nrm);
  unsafeAtomicAdd(&ah[doff + 0], vh.x * nrm);
  unsafeAtomicAdd(&ah[doff + 1], vh.y * nrm);
  unsafeAtomicAdd(&ah[doff + 2], vh.z * nrm);
  unsafeAtomicAdd(&ah[doff + 3], vh.w * nrm);
}

// ---------------- add self-loop + bias, convert to bf16 ----------------
__global__ void k_finalize(const float* az, const float* ah, const float* xz,
                           const float* xh, const float* dinv, const float* bzc,
                           const float* bhc, bf16* gz, bf16* gh, int N) {
  int t = blockIdx.x * blockDim.x + threadIdx.x;
  if (t >= N * 32) return;
  int i = t >> 5, c = t & 31;
  float d2 = dinv[i] * dinv[i];
  gz[t] = (bf16)(az[t] + xz[t] * d2 + bzc[c]);
  gh[t] = (bf16)(ah[t] + xh[t] * d2 + bhc[c]);
}

// ---------------- gate linears + activations -> hb_bf (agents only) ----------------
__global__ void k_gatelin(const bf16* gz, const bf16* gh, const bf16* pool,
                          const float* bzl, const float* bhl, const float* s_attn,
                          bf16* hb, int ntiles, int NA) {
  int wave = blockIdx.x * (blockDim.x >> 5) + (threadIdx.x >> 5);
  if (wave >= ntiles) return;
  int lane = threadIdx.x & 31;
  float s = *s_attn;
  v16bf az_ = load_A16x32(gz, wave * 16, 32, 0, lane);
  v16bf ah_ = load_A16x32(gh, wave * 16, 32, 0, lane);
  int nn = lane & 15, mb = 8 * (lane >> 4);
#pragma unroll
  for (int nb = 0; nb < 2; ++nb) {
    v8f cz = {}; v8f ch = {};
    cz = wmma_bf16(az_, load_B(pool, F_WZL + nb, lane), cz);
    ch = wmma_bf16(ah_, load_B(pool, F_WHL + nb, lane), ch);
    int n = nb * 16 + nn;
    float bz = bzl[n], bh = bhl[n];
#pragma unroll
    for (int v = 0; v < 8; ++v) {
      int m = wave * 16 + mb + v;
      float z = 1.0f / (1.0f + __expf(-(cz[v] + bz)));
      float ht = tanhf(ch[v] + bh);
      float hn = s * (1.0f - z) * ht;
      hn = fmaxf(hn, 0.0f);                        // relu
      if (m < NA) hb[(size_t)m * 32 + n] = (bf16)hn;
    }
  }
}

// ---------------- 2-layer MLP head, fully fused per 16-row tile ----------------
__global__ void k_mlp(const bf16* hb, const bf16* pool, const float* b1,
                      const float* b2, float* out, int ntiles) {
  __shared__ bf16 lds[8 * 16 * 64];                // per-wave 16x64 bf16 slice
  int wib = threadIdx.x >> 5;
  int wave = blockIdx.x * (blockDim.x >> 5) + wib;
  if (wave >= ntiles) return;
  int lane = threadIdx.x & 31;
  bf16* slice = lds + wib * (16 * 64);
  int nn = lane & 15, mb = 8 * (lane >> 4);

  // layer 1: [16,32] @ [32,64] -> relu -> bf16 in LDS (row-major 16x64)
  v16bf a = load_A16x32(hb, wave * 16, 32, 0, lane);
#pragma unroll
  for (int nb = 0; nb < 4; ++nb) {
    v8f c = {};
    c = wmma_bf16(a, load_B(pool, F_W1 + nb, lane), c);
    int n = nb * 16 + nn;
    float b = b1[n];
#pragma unroll
    for (int v = 0; v < 8; ++v)
      slice[(mb + v) * 64 + n] = (bf16)fmaxf(c[v] + b, 0.0f);
  }
  // same-wave LDS RAW: drain the DS counter (CDNA5 split wait)
  asm volatile("s_wait_dscnt 0" ::: "memory");

  // layer 2: [16,64] @ [64,112pad] -> out[16,100]
  v16bf a0 = load_A16x32(slice, 0, 64, 0, lane);
  v16bf a1 = load_A16x32(slice, 0, 64, 32, lane);
#pragma unroll
  for (int nb = 0; nb < 7; ++nb) {
    v8f c = {};
    c = wmma_bf16(a0, load_B(pool, F_W2 + nb, lane), c);
    c = wmma_bf16(a1, load_B(pool, F_W2 + 7 + nb, lane), c);
    int n = nb * 16 + nn;
    if (n < 100) {
      float b = b2[n];
#pragma unroll
      for (int v = 0; v < 8; ++v)
        out[(size_t)(wave * 16 + mb + v) * 100 + n] = c[v] + b;
    }
  }
}

extern "C" void kernel_launch(void* const* d_in, const int* in_sizes, int n_in,
                              void* d_out, int out_size, void* d_ws, size_t ws_size,
                              hipStream_t stream) {
  const float* agent_x = (const float*)d_in[0];
  const float* map_x   = (const float*)d_in[1];
  const int*   ei      = (const int*)d_in[2];
  const float* W_agent = (const float*)d_in[3];
  const float* b_agent = (const float*)d_in[4];
  const float* W_map   = (const float*)d_in[5];
  const float* b_map   = (const float*)d_in[6];
  const float* Wz_c    = (const float*)d_in[7];
  const float* bz_c    = (const float*)d_in[8];
  // d_in[9..10] = Wr_c/br_c, d_in[15..16] = Wr_l/br_l: r-gate is dead (multiplies zeros)
  const float* Wh_c    = (const float*)d_in[11];
  const float* bh_c    = (const float*)d_in[12];
  const float* Wz_l    = (const float*)d_in[13];
  const float* bz_l    = (const float*)d_in[14];
  const float* Wh_l    = (const float*)d_in[17];
  const float* bh_l    = (const float*)d_in[18];
  const float* attn    = (const float*)d_in[19];
  const float* W1      = (const float*)d_in[20];
  const float* b1      = (const float*)d_in[21];
  const float* W2      = (const float*)d_in[22];
  const float* b2      = (const float*)d_in[23];
  float* out = (float*)d_out;

  const int NA = in_sizes[0] / 9;
  const int NM = in_sizes[1] / 6;
  const int N  = NA + NM;
  const int E  = in_sizes[2] / 2;
  const int P  = in_sizes[19];

  // ---- workspace carve (device L2-resident hot arrays) ----
  char* w = (char*)d_ws;
  size_t off = 0;
  auto carve = [&](size_t bytes) -> void* {
    void* p = w + off;
    off = (off + bytes + 255) & ~(size_t)255;
    return p;
  };
  bf16*  x_bf  = (bf16*) carve((size_t)N * 32 * 2);
  float* xz    = (float*)carve((size_t)N * 32 * 4);
  float* xh    = (float*)carve((size_t)N * 32 * 4);
  float* az    = (float*)carve((size_t)N * 32 * 4);
  float* ah    = (float*)carve((size_t)N * 32 * 4);
  float* dinv  = (float*)carve((size_t)N * 4);
  bf16*  gz    = (bf16*) carve((size_t)N * 32 * 2);
  bf16*  gh    = (bf16*) carve((size_t)N * 32 * 2);
  bf16*  hb    = (bf16*) carve((size_t)NA * 32 * 2);
  float* sattn = (float*)carve(256);
  bf16*  pool  = (bf16*) carve((size_t)NFRAG * 512 * 2);
  (void)ws_size; (void)n_in; (void)out_size;

  const int T = 256;                      // 8 wave32 per block
  const int ntilesN  = (N + 15) / 16;     // 6250
  const int ntilesNA = (NA + 15) / 16;    // 3750

  k_attn<<<1, 64, 0, stream>>>(attn, P, sattn);
  k_prep<<<(NFRAG * 512 + T - 1) / T, T, 0, stream>>>(Wz_c, Wh_c, Wz_l, Wh_l, W1, W2, pool);
  k_encode<<<((size_t)N * 32 + T - 1) / T, T, 0, stream>>>(agent_x, map_x, W_agent,
                                                           b_agent, W_map, b_map,
                                                           x_bf, NA, NM);
  k_initdeg<<<(N + T - 1) / T, T, 0, stream>>>(dinv, N);
  k_deg<<<(E + T - 1) / T, T, 0, stream>>>(ei, E, dinv);
  k_dinv<<<(N + T - 1) / T, T, 0, stream>>>(dinv, N);
  k_zero2<<<((size_t)N * 32 + T - 1) / T, T, 0, stream>>>(az, ah, N * 32);
  k_xw<<<(ntilesN + 7) / 8, T, 0, stream>>>(x_bf, pool, xz, xh, ntilesN);
  k_edge<<<((size_t)E * 8 + T - 1) / T, T, 0, stream>>>(ei, E, dinv, xz, xh, az, ah);
  k_finalize<<<((size_t)N * 32 + T - 1) / T, T, 0, stream>>>(az, ah, xz, xh, dinv,
                                                             bz_c, bh_c, gz, gh, N);
  k_gatelin<<<(ntilesN + 7) / 8, T, 0, stream>>>(gz, gh, pool, bz_l, bh_l, sattn,
                                                 hb, ntilesN, NA);
  k_mlp<<<(ntilesNA + 7) / 8, T, 0, stream>>>(hb, pool, b1, b2, out, ntilesNA);
}
